// HorizonTemporalSelfAttention_82884278879239
// MI455X (gfx1250) — matrix operand order repaired and checked by reading, not compiled, not run on hardware
//
#include <hip/hip_runtime.h>
#include <hip/hip_bf16.h>

typedef __attribute__((ext_vector_type(16))) _Float16 v16h;
typedef __attribute__((ext_vector_type(8)))  _Float16 v8h;
typedef __attribute__((ext_vector_type(8)))  float    v8f;

#define NHD   8
#define NLV   4
#define NPT   4
#define NQ    16384
#define BSZ   2
#define EMB   256
#define NCOL  384          // 256 offset cols + 128 attn cols
#define TOTAL 21760        // 128*128 + 64*64 + 32*32 + 16*16

// ---------- helpers ----------
__device__ __forceinline__ float bperm(float v, int srcLane) {
    return __builtin_bit_cast(float,
        __builtin_amdgcn_ds_bpermute(srcLane << 2, __builtin_bit_cast(int, v)));
}

__device__ __forceinline__ v8h cvt8(float4 a, float4 b) {
    v8h r;
    r[0] = (_Float16)a.x; r[1] = (_Float16)a.y; r[2] = (_Float16)a.z; r[3] = (_Float16)a.w;
    r[4] = (_Float16)b.x; r[5] = (_Float16)b.y; r[6] = (_Float16)b.z; r[7] = (_Float16)b.w;
    return r;
}

__device__ __forceinline__ v16h cat16(v8h lo, v8h hi) {
    v16h r;
#pragma unroll
    for (int i = 0; i < 8; ++i) { r[i] = lo[i]; r[i + 8] = hi[i]; }
    return r;
}

// =====================================================================
// Kernel 1: P[m, 0:384] = query[m, :] @ [W_off; W_attn]^T + [b_off; b_attn]
// M = BSZ*NQ = 32768, K = 256, N = 384.  f16 WMMA, f32 accumulate.
// grid = (3 col-groups, 256 m-chunks), block = 256 (8 waves).
// Wave w owns N-tile n0 = 128*gx + 16*w, holds its 8 B-fragments in
// registers, and loops 8 M-tiles whose A tile is staged in LDS.
// =====================================================================
__global__ __launch_bounds__(256) void msda_proj_wmma(
    const float* __restrict__ query,
    const float* __restrict__ W_off,  const float* __restrict__ b_off,
    const float* __restrict__ W_attn, const float* __restrict__ b_attn,
    float* __restrict__ P)
{
    __shared__ _Float16 smA[16][264];   // 16 rows x 256 halfs, +8 pad (bank spread)

    const int t    = threadIdx.x;
    const int lane = t & 31;
    const int w    = t >> 5;

    const int n0 = blockIdx.x * 128 + w * 16;      // wave's N-tile base
    const int nn = n0 + (lane & 15);               // this lane's output column

    const float* wrow;
    float bias;
    if (n0 < 256) { wrow = W_off  + (size_t)nn * 256;         bias = b_off[nn]; }
    else          { wrow = W_attn + (size_t)(nn - 256) * 256; bias = b_attn[nn - 256]; }

    // ---- B fragments: B(k,n), n = lane%16, k = (lane/16)*16 + h + 32*ks ----
    v16h bf[8];
    {
        const int kbb = (lane >> 4) << 4;   // 0 or 16
#pragma unroll
        for (int ks = 0; ks < 8; ++ks) {
            const float4* src = (const float4*)(wrow + ks * 32 + kbb);
            float4 f0 = src[0], f1 = src[1], f2 = src[2], f3 = src[3];
            bf[ks] = cat16(cvt8(f0, f1), cvt8(f2, f3));
        }
    }

    const int mrow_half = (lane >> 4) << 3;   // 0 or 8: D rows 0-7 vs 8-15
    const int mm        = lane & 15;          // A row within tile
    const int kba       = (lane >> 4) << 4;   // A frag k base: 0 or 16

    const int mt0 = blockIdx.y * 8;
    for (int mi = 0; mi < 8; ++mi) {
        const int mbase = (mt0 + mi) * 16;

        __syncthreads();   // protect previous iteration's LDS reads
        {
            // cooperative load of A tile 16x256 fp32 -> f16 LDS
            const int row = t >> 4;
            const int seg = t & 15;
            const float4* src =
                (const float4*)(query + (size_t)(mbase + row) * 256 + seg * 16);
            float4 f0 = src[0], f1 = src[1], f2 = src[2], f3 = src[3];
            *(v8h*)(&smA[row][seg * 16])     = cvt8(f0, f1);
            *(v8h*)(&smA[row][seg * 16 + 8]) = cvt8(f2, f3);
        }
        __syncthreads();

        v8f acc;
#pragma unroll
        for (int r = 0; r < 8; ++r) acc[r] = bias;

#pragma unroll
        for (int ks = 0; ks < 8; ++ks) {
            // A(m,k): m = lane%16, k = 32*ks + kba + h
            v8h lo = *(const v8h*)(&smA[mm][ks * 32 + kba]);
            v8h hi = *(const v8h*)(&smA[mm][ks * 32 + kba + 8]);
            v16h af = cat16(lo, hi);
            acc = __builtin_amdgcn_wmma_f32_16x16x32_f16(
                false, af, false, bf[ks], (short)0, acc, false, false);
        }

        // D(m,n): lane = n + 16*(m/8), VGPR r = m%8
        float* dst = P + (size_t)(mbase + mrow_half) * NCOL + n0 + (lane & 15);
#pragma unroll
        for (int r = 0; r < 8; ++r) dst[(size_t)r * NCOL] = acc[r];
    }
}

// =====================================================================
// Kernel 2: softmax + bilinear sampling + reduction.
// block = 256 threads = 8 waves, covers one (b,h) and 32 queries.
// lane = channel (d = 32), each wave handles 4 queries sequentially.
// grid = (NQ/32, BSZ*NHD)
// =====================================================================
__global__ __launch_bounds__(256) void msda_sample(
    const float* __restrict__ P,
    const float* __restrict__ value,
    const float* __restrict__ refp,
    float* __restrict__ out)
{
    __shared__ float tr[32][33];   // [channel][query-slot], padded

    const int t     = threadIdx.x;
    const int lane  = t & 31;
    const int w     = t >> 5;
    const int b     = blockIdx.y >> 3;
    const int h     = blockIdx.y & 7;
    const int qbase = blockIdx.x * 32;

    // per-lane channel pointer into value[b, :, h, lane]
    const float* vbase = value + (size_t)b * TOTAL * 256 + h * 32 + lane;

    constexpr int cW[4] = {128, 64, 32, 16};
    constexpr int cH[4] = {128, 64, 32, 16};
    constexpr int cS[4] = {0, 16384, 20480, 21504};

    for (int qi = 0; qi < 4; ++qi) {
        const int    q   = qbase + w * 4 + qi;
        const size_t row = (size_t)b * NQ + q;

        const float offv = P[row * NCOL + h * 32 + lane];              // (l,p,xy) flat
        const float av   = P[row * NCOL + 256 + h * 16 + (lane & 15)]; // (l,p) flat
        const float rv   = refp[row * 8 + (lane & 7)];                 // 8 ref coords

        // softmax over 16 logits (lanes 16-31 mirror 0-15): xor butterfly
        float mx = av;
#pragma unroll
        for (int d2 = 1; d2 < 16; d2 <<= 1) mx = fmaxf(mx, bperm(mx, lane ^ d2));
        float ex = __expf(av - mx);
        float sm = ex;
#pragma unroll
        for (int d2 = 1; d2 < 16; d2 <<= 1) sm += bperm(sm, lane ^ d2);
        const float smv = ex / sm;

        float acc = 0.f;
#pragma unroll
        for (int sp = 0; sp < 16; ++sp) {
            const int l = sp >> 2;
            const int p = sp & 3;
            const float ox = bperm(offv, 2 * sp);
            const float oy = bperm(offv, 2 * sp + 1);
            // reference quirk: ref point indexed by the POINT index p, not level
            const float rx = bperm(rv, 2 * p);
            const float ry = bperm(rv, 2 * p + 1);
            const float aw = bperm(smv, sp);

            // (ref + off/W)*2-1 mapped back to pixels == ref*W + off - 0.5
            const float x = rx * (float)cW[l] + ox - 0.5f;
            const float y = ry * (float)cH[l] + oy - 0.5f;
            const float fx0 = floorf(x), fy0 = floorf(y);
            const int   ix = (int)fx0,   iy = (int)fy0;
            const float wx = x - fx0,    wy = y - fy0;

#pragma unroll
            for (int dy = 0; dy < 2; ++dy) {
                const int ty = iy + dy;
                if (ty < 0 || ty >= cH[l]) continue;       // wave-uniform branch
                const float wyv = dy ? wy : 1.f - wy;
#pragma unroll
                for (int dx = 0; dx < 2; ++dx) {
                    const int tx = ix + dx;
                    if (tx < 0 || tx >= cW[l]) continue;
                    const float wxv = dx ? wx : 1.f - wx;
                    const float v = vbase[(size_t)(cS[l] + ty * cW[l] + tx) * 256];
                    acc += aw * wyv * wxv * v;             // coalesced 128B tap
                }
            }
        }
        tr[lane][w * 4 + qi] = acc;
    }

    __syncthreads();
    // out[b, h*32+ch, q]: transpose so stores are contiguous over q
#pragma unroll
    for (int it = 0; it < 4; ++it) {
        const int idx = it * 256 + t;
        const int ch  = idx >> 5;
        const int qq  = idx & 31;
        out[((size_t)b * 256 + h * 32 + ch) * (size_t)NQ + qbase + qq] = tr[ch][qq];
    }
}

// =====================================================================
extern "C" void kernel_launch(void* const* d_in, const int* in_sizes, int n_in,
                              void* d_out, int out_size, void* d_ws, size_t ws_size,
                              hipStream_t stream) {
    (void)in_sizes; (void)n_in; (void)out_size; (void)ws_size;
    const float* query  = (const float*)d_in[0];
    const float* value  = (const float*)d_in[1];
    const float* refp   = (const float*)d_in[2];
    const float* W_off  = (const float*)d_in[3];
    const float* b_off  = (const float*)d_in[4];
    const float* W_attn = (const float*)d_in[5];
    const float* b_attn = (const float*)d_in[6];
    // d_in[7] = spatial_shapes (static; hardcoded as compile-time constants)

    float* out = (float*)d_out;
    float* P   = (float*)d_ws;   // [32768 x 384] fp32 = 48 MB

    dim3 g1(3, 256), b1(256);
    hipLaunchKernelGGL(msda_proj_wmma, g1, b1, 0, stream,
                       query, W_off, b_off, W_attn, b_attn, P);

    dim3 g2(NQ / 32, BSZ * NHD), b2(256);
    hipLaunchKernelGGL(msda_sample, g2, b2, 0, stream, P, value, refp, out);
}
